// Model_61555471287011
// MI455X (gfx1250) — matrix operand order repaired and checked by reading, not compiled, not run on hardware
//
#include <hip/hip_runtime.h>
#include <math.h>

typedef __attribute__((ext_vector_type(16))) __bf16 v16bf;
typedef __attribute__((ext_vector_type(8)))  __bf16 v8bf;
typedef __attribute__((ext_vector_type(8)))  float  v8f;

#define BSZ   4
#define NN    512
#define DD    64
#define NKEEP 256
#define WAVES 8      // waves per block in the attention kernel (1 row i per wave)

#define LOG2E  1.4426950408889634f

// Branch-free native-instruction transcendentals (v_exp_f32 / v_rcp_f32).
__device__ __forceinline__ float fast_exp(float x) {
    return __builtin_amdgcn_exp2f(x * LOG2E);
}
__device__ __forceinline__ float fast_tanh(float x) {
    // tanh(x) = 1 - 2/(e^{2x}+1); exp2 overflow->inf gives +1, underflow->0 gives -1.
    float e = __builtin_amdgcn_exp2f(x * (2.0f * LOG2E));
    return 1.0f - 2.0f * __builtin_amdgcn_rcpf(e + 1.0f);
}
__device__ __forceinline__ float fast_sigmoid(float x) {
    return __builtin_amdgcn_rcpf(1.0f + __builtin_amdgcn_exp2f(-x * LOG2E));
}

// -------------------------------------------------------------------------
// Kernel A: per (b, i) attention row via WMMA GEMM  X_b(512x64) x M_i(64x64),
// M_i = diag(x_i) * att_proj_w, then tanh/att_weight epilogue -> scores,
// softmax over j, agg = att @ x, two 64x64 GEMV heads, BN + SELU,
// pool logit sigmoid.  Writes h (BS,N,64) and s (BS,N) to workspace.
// -------------------------------------------------------------------------
__global__ __launch_bounds__(32 * WAVES, 1)
void gat_attention_kernel(const float* __restrict__ x,
                          const float* __restrict__ att_w,   // (64,64) [d][e]
                          const float* __restrict__ att_b,   // (64)
                          const float* __restrict__ att_vec, // (64)
                          const float* __restrict__ pwa_w,   // (64,64)
                          const float* __restrict__ pwa_b,
                          const float* __restrict__ pwoa_w,  // (64,64)
                          const float* __restrict__ pwoa_b,
                          const float* __restrict__ bn_g,
                          const float* __restrict__ bn_b,
                          const float* __restrict__ bn_m,
                          const float* __restrict__ bn_v,
                          const float* __restrict__ pool_w,
                          const float* __restrict__ pool_b,
                          float* __restrict__ h_out,         // (BS,N,64)
                          float* __restrict__ s_out)         // (BS,N)
{
    __shared__ __align__(32) __bf16 Xb[NN][DD];      // 64 KB  x_b as bf16, [j][d]
    __shared__ __align__(32) __bf16 Wt[DD][DD];      //  8 KB  att_proj_w^T, [e][d]
    __shared__ float sc[WAVES][NN];                  // 16 KB  scores / att per wave
    __shared__ float aggsh[WAVES][DD];               //  2 KB
    __shared__ float bias_sh[DD];
    __shared__ float aw_sh[DD];

    const int b     = blockIdx.x / (NN / WAVES);
    const int itile = blockIdx.x % (NN / WAVES);
    const int tid   = threadIdx.x;
    const int wave  = tid >> 5;
    const int lane  = tid & 31;
    const int nloc  = lane & 15;     // column within 16-wide tile
    const int khalf = lane >> 4;     // lane half

    // ---- cooperative staging ----
    const float* xb = x + (size_t)b * NN * DD;
    for (int k = tid; k < NN * DD; k += blockDim.x)
        ((__bf16*)Xb)[k] = (__bf16)xb[k];
    for (int k = tid; k < DD * DD; k += blockDim.x) {
        int d = k / DD, e = k % DD;          // att_w is [d][e] row-major
        Wt[e][d] = (__bf16)att_w[k];
    }
    if (tid < DD) { bias_sh[tid] = att_b[tid]; aw_sh[tid] = att_vec[tid]; }
    __syncthreads();

    const int i = itile * WAVES + wave;      // the row this wave owns

    // ---- build B fragments of M_i = diag(x_i) * W  (invariant over j) ----
    // B frag here (contiguous k, 32x16 bf16): lane<16 -> n=lane, K=0..15; lane>=16 -> n=lane-16, K=16..31
    // vector element t of v16bf = K offset t within the lane's contiguous 16-K range.
    v16bf Bf[2][4];
    #pragma unroll
    for (int kt = 0; kt < 2; ++kt) {
        const int kbase = kt * 32 + khalf * 16;
        const v16bf xv = *(const v16bf*)&Xb[i][kbase];   // x_i[k..k+15]
        #pragma unroll
        for (int et = 0; et < 4; ++et) {
            const int e = et * 16 + nloc;
            const v16bf wv = *(const v16bf*)&Wt[e][kbase];
            v16bf f;
            #pragma unroll
            for (int t = 0; t < 16; ++t)
                f[t] = (__bf16)((float)wv[t] * (float)xv[t]);
            Bf[kt][et] = f;
        }
    }

    // ---- main loop: 32 j-tiles, 8 WMMAs each ----
    // A frag here (contiguous k as in B, 16x32 bf16): row M = lane%16; lane<16 K elems {0..15},
    // lane>=16 {16..31} -> two 16B contiguous LDS reads per fragment.
    const int jrowbase = khalf * 16;  // K sub-offset for this lane half
    #pragma unroll 2
    for (int jt = 0; jt < NN / 16; ++jt) {
        const int jrow = jt * 16 + nloc;
        v16bf Afr[2];
        #pragma unroll
        for (int kt = 0; kt < 2; ++kt) {
            v8bf lo = *(const v8bf*)&Xb[jrow][kt * 32 + jrowbase];
            v8bf hi = *(const v8bf*)&Xb[jrow][kt * 32 + 8 + jrowbase];
            Afr[kt] = __builtin_shufflevector(lo, hi, 0, 1, 2, 3, 4, 5, 6, 7,
                                              8, 9, 10, 11, 12, 13, 14, 15);
        }
        float part[8] = {0.f, 0.f, 0.f, 0.f, 0.f, 0.f, 0.f, 0.f};
        #pragma unroll
        for (int et = 0; et < 4; ++et) {
            v8f c = {};
            c = __builtin_amdgcn_wmma_f32_16x16x32_bf16(false, Afr[0], false, Bf[0][et],
                                                        (short)0, c, false, false);
            c = __builtin_amdgcn_wmma_f32_16x16x32_bf16(false, Afr[1], false, Bf[1][et],
                                                        (short)0, c, false, false);
            const int e  = et * 16 + nloc;       // C col N = lane%16
            const float aw = aw_sh[e];
            const float bb = bias_sh[e];
            #pragma unroll
            for (int v = 0; v < 8; ++v)          // C row M = v + 8*khalf
                part[v] += aw * fast_tanh(c[v] + bb);
        }
        // reduce over e: sum across the 16 lanes of each half
        #pragma unroll
        for (int v = 0; v < 8; ++v) {
            float p = part[v];
            p += __shfl_xor(p, 1, 16);
            p += __shfl_xor(p, 2, 16);
            p += __shfl_xor(p, 4, 16);
            p += __shfl_xor(p, 8, 16);
            part[v] = p;
        }
        if (nloc == 0) {
            #pragma unroll
            for (int v = 0; v < 8; ++v)
                sc[wave][jt * 16 + khalf * 8 + v] = part[v];
        }
    }

    // ---- softmax over j (512 values, 16 per lane) ----
    float sv[16];
    float mx = -INFINITY;
    #pragma unroll
    for (int t = 0; t < 16; ++t) { sv[t] = sc[wave][lane + 32 * t]; mx = fmaxf(mx, sv[t]); }
    #pragma unroll
    for (int m = 16; m >= 1; m >>= 1) mx = fmaxf(mx, __shfl_xor(mx, m, 32));
    float sum = 0.f;
    #pragma unroll
    for (int t = 0; t < 16; ++t) { sv[t] = fast_exp(sv[t] - mx); sum += sv[t]; }
    #pragma unroll
    for (int m = 16; m >= 1; m >>= 1) sum += __shfl_xor(sum, m, 32);
    const float inv = __builtin_amdgcn_rcpf(sum);
    #pragma unroll
    for (int t = 0; t < 16; ++t) sc[wave][lane + 32 * t] = sv[t] * inv;

    // ---- agg[d] = sum_j att[j] * x_b[j][d]   (2 d-columns per lane) ----
    const int d0 = lane, d1 = lane + 32;
    float a0 = 0.f, a1 = 0.f;
    for (int j = 0; j < NN; ++j) {
        const float w = sc[wave][j];
        a0 += w * (float)Xb[j][d0];
        a1 += w * (float)Xb[j][d1];
    }
    aggsh[wave][d0] = a0;
    aggsh[wave][d1] = a1;

    // ---- h = agg@pwa_w + pwa_b + x_i@pwoa_w + pwoa_b ; BN ; SELU ----
    const float* xrow = x + ((size_t)b * NN + i) * DD;   // fp32 x_i
    const int e0 = 2 * lane, e1 = 2 * lane + 1;
    float h0 = pwa_b[e0] + pwoa_b[e0];
    float h1 = pwa_b[e1] + pwoa_b[e1];
    for (int d = 0; d < DD; ++d) {
        const float ag = aggsh[wave][d];
        const float xv = xrow[d];
        h0 += ag * pwa_w[d * DD + e0] + xv * pwoa_w[d * DD + e0];
        h1 += ag * pwa_w[d * DD + e1] + xv * pwoa_w[d * DD + e1];
    }
    const float selu_a = 1.6732632423543772f, selu_s = 1.0507009873554805f;
    {
        float t0 = (h0 - bn_m[e0]) * rsqrtf(bn_v[e0] + 1e-5f) * bn_g[e0] + bn_b[e0];
        float t1 = (h1 - bn_m[e1]) * rsqrtf(bn_v[e1] + 1e-5f) * bn_g[e1] + bn_b[e1];
        h0 = selu_s * (t0 > 0.f ? t0 : selu_a * (fast_exp(t0) - 1.f));
        h1 = selu_s * (t1 > 0.f ? t1 : selu_a * (fast_exp(t1) - 1.f));
    }
    float* hrow = h_out + ((size_t)b * NN + i) * DD;
    hrow[e0] = h0;
    hrow[e1] = h1;

    // ---- pool logit -> sigmoid ----
    float ps = h0 * pool_w[e0] + h1 * pool_w[e1];
    #pragma unroll
    for (int m = 16; m >= 1; m >>= 1) ps += __shfl_xor(ps, m, 32);
    if (lane == 0)
        s_out[(size_t)b * NN + i] = fast_sigmoid(ps + pool_b[0]);
}

// -------------------------------------------------------------------------
// Kernel B: exact torch.topk(desc, ties by index) via O(N^2) ranking,
// then gather h*s rows into output slot = rank.
// -------------------------------------------------------------------------
__global__ __launch_bounds__(256)
void gat_pool_kernel(const float* __restrict__ h,   // (BS,N,64)
                     const float* __restrict__ s,   // (BS,N)
                     float* __restrict__ out)       // (BS,NKEEP,64)
{
    __shared__ float ssh[NN];
    const int b   = blockIdx.x;
    const int tid = threadIdx.x;
    for (int k = tid; k < NN; k += blockDim.x) ssh[k] = s[(size_t)b * NN + k];
    __syncthreads();

    for (int i = tid; i < NN; i += blockDim.x) {
        const float si = ssh[i];
        int rank = 0;
        for (int j = 0; j < NN; ++j) {
            const float sj = ssh[j];
            rank += (sj > si) || (sj == si && j < i);
        }
        if (rank < NKEEP) {
            const float* hr = h + ((size_t)b * NN + i) * DD;
            float* orow = out + ((size_t)b * NKEEP + rank) * DD;
            for (int e = 0; e < DD; ++e) orow[e] = hr[e] * si;
        }
    }
}

// -------------------------------------------------------------------------
extern "C" void kernel_launch(void* const* d_in, const int* in_sizes, int n_in,
                              void* d_out, int out_size, void* d_ws, size_t ws_size,
                              hipStream_t stream) {
    const float* x       = (const float*)d_in[0];
    const float* att_w   = (const float*)d_in[1];
    const float* att_b   = (const float*)d_in[2];
    const float* att_vec = (const float*)d_in[3];
    const float* pwa_w   = (const float*)d_in[4];
    const float* pwa_b   = (const float*)d_in[5];
    const float* pwoa_w  = (const float*)d_in[6];
    const float* pwoa_b  = (const float*)d_in[7];
    const float* bn_g    = (const float*)d_in[8];
    const float* bn_b    = (const float*)d_in[9];
    const float* bn_m    = (const float*)d_in[10];
    const float* bn_v    = (const float*)d_in[11];
    const float* pool_w  = (const float*)d_in[12];
    const float* pool_b  = (const float*)d_in[13];

    float* h_ws = (float*)d_ws;                         // BS*N*64 floats
    float* s_ws = h_ws + (size_t)BSZ * NN * DD;         // BS*N floats
    float* out  = (float*)d_out;                        // BS*NKEEP*64 floats

    dim3 gridA(BSZ * (NN / WAVES));
    dim3 blockA(32 * WAVES);
    gat_attention_kernel<<<gridA, blockA, 0, stream>>>(
        x, att_w, att_b, att_vec, pwa_w, pwa_b, pwoa_w, pwoa_b,
        bn_g, bn_b, bn_m, bn_v, pool_w, pool_b, h_ws, s_ws);

    gat_pool_kernel<<<BSZ, 256, 0, stream>>>(h_ws, s_ws, out);
}
